// EdgeGenerator_26628797235676
// MI455X (gfx1250) — matrix operand order repaired and checked by reading, not compiled, not run on hardware
//
#include <hip/hip_runtime.h>

typedef __bf16 bf16;
typedef __attribute__((ext_vector_type(16))) __bf16 v16bf;
typedef __attribute__((ext_vector_type(8)))  __bf16 v8bf;
typedef __attribute__((ext_vector_type(8)))  float   v8f;

#define NN_   10000
#define MP_   10048   /* NN_ padded to a multiple of the 64-row GEMM block tile */
#define ND_   128
#define HID_  256
#define NE_   320000
#define NL_   3

// ---------- helpers ----------
__device__ __forceinline__ bf16 f2b(float f) {
  unsigned u = __builtin_bit_cast(unsigned, f);
  unsigned r = (u + 0x7fffu + ((u >> 16) & 1u)) >> 16;
  unsigned short s = (unsigned short)r;
  return __builtin_bit_cast(bf16, s);
}
__device__ __forceinline__ float wave_sum(float v) {
  for (int o = 16; o > 0; o >>= 1) v += __shfl_xor(v, o, 32);
  return v;
}
__device__ __forceinline__ float silu_f(float x) { return x / (1.f + __expf(-x)); }

// Assemble a 16x32 A fragment from two 16B runs (contiguous k order, matches B):
// lane L: M=L%16, runs at k0+(L/16)*16 and k0+8+(L/16)*16.
__device__ __forceinline__ v16bf frag_from_runs(const bf16* p) {
  v8bf lo = *reinterpret_cast<const v8bf*>(p);
  v8bf hi = *reinterpret_cast<const v8bf*>(p + 8);
  return __builtin_shufflevector(lo, hi, 0,1,2,3,4,5,6,7,8,9,10,11,12,13,14,15);
}

// ---------- elementwise / prep kernels ----------
__global__ void eg_f2bf(const float* in, bf16* out, int n) {
  int i = blockIdx.x * blockDim.x + threadIdx.x;
  if (i < n) out[i] = f2b(in[i]);
}
__global__ void eg_zero_bf(bf16* p, int n) {
  int i = blockIdx.x * blockDim.x + threadIdx.x;
  if (i < n) p[i] = f2b(0.f);
}
// out[n*K+k] = bf16(in[k*N+n])  (W[K][N] -> Wt[N][K] in bf16)
__global__ void eg_transpose_bf(const float* in, bf16* out, int K, int N) {
  int i = blockIdx.x * blockDim.x + threadIdx.x;
  if (i >= K * N) return;
  int n = i / K, k = i % K;
  out[i] = f2b(in[(size_t)k * N + n]);
}
__global__ void eg_fill1(float* p, int n) {
  int i = blockIdx.x * blockDim.x + threadIdx.x;
  if (i < n) p[i] = 1.0f;   // self-loop contribution to degree (pad rows stay 1)
}
__global__ void eg_deg(const int* dst, float* deg, int e) {
  int i = blockIdx.x * blockDim.x + threadIdx.x;
  if (i < e) atomicAdd(&deg[dst[i]], 1.0f);
}
__global__ void eg_rsqrt(float* p, int n) {
  int i = blockIdx.x * blockDim.x + threadIdx.x;
  if (i < n) p[i] = rsqrtf(p[i]);
}

// ---------- WMMA GEMM: C[MP][256](+bias) = A(bf16,MPxK) @ Bt(bf16,256xK)^T ----------
// 128 threads = 4 waves (2x2), block tile 64x64, wave tile 32x32, no bounds checks
// (M padded to a multiple of 64; pad rows are never consumed by real outputs).
template <int K, bool HAS_BIAS, bool WF32, bool WBF16>
__global__ __launch_bounds__(128) void eg_wmma_gemm(
    const bf16* __restrict__ A, const bf16* __restrict__ Bt,
    const float* __restrict__ bias, float* __restrict__ C,
    bf16* __restrict__ Cb) {
  constexpr int N = HID_;
  int lane = threadIdx.x & 31;
  int wave = threadIdx.x >> 5;
  int bm = blockIdx.y * 64 + (wave >> 1) * 32;
  int bn = blockIdx.x * 64 + (wave & 1) * 32;

  const bf16* a0p = A  + (size_t)(bm + (lane & 15)) * K + ((lane >> 4) << 4);
  const bf16* a1p = a0p + (size_t)16 * K;
  const bf16* b0p = Bt + (size_t)(bn + (lane & 15)) * K + ((lane >> 4) << 4);
  const bf16* b1p = b0p + (size_t)16 * K;

  v8f acc[2][2] = {};
#pragma unroll
  for (int k0 = 0; k0 < K; k0 += 32) {
    v16bf a0 = frag_from_runs(a0p + k0);
    v16bf a1 = frag_from_runs(a1p + k0);
    v16bf b0 = *reinterpret_cast<const v16bf*>(b0p + k0);
    v16bf b1 = *reinterpret_cast<const v16bf*>(b1p + k0);
    acc[0][0] = __builtin_amdgcn_wmma_f32_16x16x32_bf16(false, a0, false, b0, (short)0, acc[0][0], false, false);
    acc[0][1] = __builtin_amdgcn_wmma_f32_16x16x32_bf16(false, a0, false, b1, (short)0, acc[0][1], false, false);
    acc[1][0] = __builtin_amdgcn_wmma_f32_16x16x32_bf16(false, a1, false, b0, (short)0, acc[1][0], false, false);
    acc[1][1] = __builtin_amdgcn_wmma_f32_16x16x32_bf16(false, a1, false, b1, (short)0, acc[1][1], false, false);
  }

  int rbase = (lane >> 4) * 8;
  int col   = lane & 15;
#pragma unroll
  for (int j = 0; j < 2; ++j) {
    int c = bn + j * 16 + col;
    float bv = HAS_BIAS ? bias[c] : 0.f;
#pragma unroll
    for (int i = 0; i < 2; ++i) {
#pragma unroll
      for (int r = 0; r < 8; ++r) {
        int m = bm + i * 16 + rbase + r;
        float v = acc[i][j][r] + bv;
        if (WF32)  C[(size_t)m * N + c]  = v;
        if (WBF16) Cb[(size_t)m * N + c] = f2b(v);
      }
    }
  }
}

// ---------- encoder LayerNorm + SiLU -> bf16 (one wave per 256-wide row) ----------
__global__ __launch_bounds__(256) void eg_ln_silu(
    const float* __restrict__ X, const float* __restrict__ g,
    const float* __restrict__ b, bf16* __restrict__ Y, int M) {
  int row  = blockIdx.x * 8 + (threadIdx.x >> 5);
  int lane = threadIdx.x & 31;
  if (row >= M) return;
  const float* x = X + (size_t)row * HID_;
  float v[8]; float s = 0.f;
#pragma unroll
  for (int j = 0; j < 8; ++j) { v[j] = x[lane + 32 * j]; s += v[j]; }
  float mean = wave_sum(s) * (1.f / HID_);
  float q = 0.f;
#pragma unroll
  for (int j = 0; j < 8; ++j) { float d = v[j] - mean; q += d * d; }
  float rstd = rsqrtf(wave_sum(q) * (1.f / HID_) + 1e-5f);
#pragma unroll
  for (int j = 0; j < 8; ++j) {
    int c = lane + 32 * j;
    Y[(size_t)row * HID_ + c] = f2b(silu_f((v[j] - mean) * rstd * g[c] + b[c]));
  }
}

// ---------- GCN aggregation ----------
__global__ void eg_agg_init(const float* xw, const float* dinv, float* agg, int n) {
  int i = blockIdx.x * blockDim.x + threadIdx.x;
  if (i >= n) return;
  float di = dinv[i / HID_];
  agg[i] = xw[i] * di * di;            // self-loop term
}
__global__ void eg_agg_edges(const float* __restrict__ xw, const int* __restrict__ src,
                             const int* __restrict__ dst, const float* __restrict__ dinv,
                             float* __restrict__ agg, int e) {
  int t = blockIdx.x * blockDim.x + threadIdx.x;
  if (t >= e * 32) return;
  int ei = t >> 5, c0 = (t & 31) * 8;
  int s = src[ei], d = dst[ei];
  float nm = dinv[s] * dinv[d];
  const float* xp = xw  + (size_t)s * HID_ + c0;
  float*       ap = agg + (size_t)d * HID_ + c0;
#pragma unroll
  for (int j = 0; j < 8; ++j) atomicAdd(ap + j, xp[j] * nm);
}
__global__ void eg_finalize(const float* agg, const float* bias, bf16* h, int n) {
  int i = blockIdx.x * blockDim.x + threadIdx.x;
  if (i >= n) return;
  h[i] = f2b(silu_f(agg[i] + bias[i % HID_]));
}

// ---------- fused edge head: gathered GEMM(16x512x256) + LN + SiLU + dot + sigmoid ----------
__global__ __launch_bounds__(256) void eg_edge_head(
    const bf16* __restrict__ H, const int* __restrict__ src, const int* __restrict__ dst,
    const bf16* __restrict__ W1t, const float* __restrict__ b1,
    const float* __restrict__ lng, const float* __restrict__ lnb,
    const float* __restrict__ w2, const float* __restrict__ b2,
    float* __restrict__ out) {
  __shared__ float tile[16][257];       // stride 257: half-wave rows hit distinct banks
  int lane = threadIdx.x & 31, wave = threadIdx.x >> 5;  // 8 waves cover 256 cols
  int e0 = blockIdx.x * 16;
  int m  = lane & 15;
  int e  = e0 + m;                       // NE_ is a multiple of 16
  const bf16* ps = H + (size_t)src[e] * HID_;
  const bf16* pd = H + (size_t)dst[e] * HID_;
  int kb = (lane >> 4) << 4;
  int n0 = wave * 32;
  const bf16* b0p = W1t + (size_t)(n0 + m) * (2 * HID_) + ((lane >> 4) << 4);
  const bf16* b1p = b0p + (size_t)16 * (2 * HID_);
  v8f acc0 = {}, acc1 = {};
#pragma unroll
  for (int k0 = 0; k0 < 2 * HID_; k0 += 32) {
    int c1 = k0 + kb, c2 = k0 + 8 + kb;    // 8-elem runs never cross the src/dst split
    v8bf lo = *reinterpret_cast<const v8bf*>(c1 < HID_ ? ps + c1 : pd + (c1 - HID_));
    v8bf hi = *reinterpret_cast<const v8bf*>(c2 < HID_ ? ps + c2 : pd + (c2 - HID_));
    v16bf a = __builtin_shufflevector(lo, hi, 0,1,2,3,4,5,6,7,8,9,10,11,12,13,14,15);
    v16bf bb0 = *reinterpret_cast<const v16bf*>(b0p + k0);
    v16bf bb1 = *reinterpret_cast<const v16bf*>(b1p + k0);
    acc0 = __builtin_amdgcn_wmma_f32_16x16x32_bf16(false, a, false, bb0, (short)0, acc0, false, false);
    acc1 = __builtin_amdgcn_wmma_f32_16x16x32_bf16(false, a, false, bb1, (short)0, acc1, false, false);
  }
  int rbase = (lane >> 4) * 8, col = lane & 15;
#pragma unroll
  for (int r = 0; r < 8; ++r) {
    int ca = n0 + col, cb = n0 + 16 + col;
    tile[rbase + r][ca] = acc0[r] + b1[ca];
    tile[rbase + r][cb] = acc1[r] + b1[cb];
  }
  __syncthreads();
#pragma unroll
  for (int rr = 0; rr < 2; ++rr) {
    int row = wave * 2 + rr;
    float v[8]; float s = 0.f;
#pragma unroll
    for (int j = 0; j < 8; ++j) { v[j] = tile[row][lane + 32 * j]; s += v[j]; }
    float mean = wave_sum(s) * (1.f / HID_);
    float q = 0.f;
#pragma unroll
    for (int j = 0; j < 8; ++j) { float d = v[j] - mean; q += d * d; }
    float rstd = rsqrtf(wave_sum(q) * (1.f / HID_) + 1e-5f);
    float dotp = 0.f;
#pragma unroll
    for (int j = 0; j < 8; ++j) {
      int c = lane + 32 * j;
      dotp += silu_f((v[j] - mean) * rstd * lng[c] + lnb[c]) * w2[c];
    }
    dotp = wave_sum(dotp);
    if (lane == 0) out[e0 + row] = 1.f / (1.f + __expf(-(dotp + b2[0])));
  }
}

// ---------- launch ----------
extern "C" void kernel_launch(void* const* d_in, const int* in_sizes, int n_in,
                              void* d_out, int out_size, void* d_ws, size_t ws_size,
                              hipStream_t stream) {
  (void)in_sizes; (void)n_in; (void)out_size; (void)ws_size;
  const float* nf      = (const float*)d_in[0];
  const int*   ei      = (const int*)d_in[1];
  const float* enc_w1  = (const float*)d_in[2];
  const float* enc_b1  = (const float*)d_in[3];
  const float* enc_lng = (const float*)d_in[4];
  const float* enc_lnb = (const float*)d_in[5];
  const float* enc_w2  = (const float*)d_in[6];
  const float* enc_b2  = (const float*)d_in[7];
  const float* gnn_w   = (const float*)d_in[8];
  const float* gnn_b   = (const float*)d_in[9];
  const float* edge_w1 = (const float*)d_in[10];
  const float* edge_b1 = (const float*)d_in[11];
  const float* e_lng   = (const float*)d_in[12];
  const float* e_lnb   = (const float*)d_in[13];
  const float* edge_w2 = (const float*)d_in[14];
  const float* edge_b2 = (const float*)d_in[15];
  const int* src = ei;
  const int* dst = ei + NE_;
  float* out = (float*)d_out;

  char* p = (char*)d_ws;
  auto alloc = [&](size_t bytes) { char* r = p; p += (bytes + 255) & ~(size_t)255; return (void*)r; };
  bf16*  nf_bf = (bf16*)alloc((size_t)MP_ * ND_ * 2);
  bf16*  w1T   = (bf16*)alloc((size_t)HID_ * ND_ * 2);
  bf16*  w2T   = (bf16*)alloc((size_t)HID_ * HID_ * 2);
  bf16*  gT    = (bf16*)alloc((size_t)NL_ * HID_ * HID_ * 2);
  bf16*  e1T   = (bf16*)alloc((size_t)HID_ * 2 * HID_ * 2);
  float* t1    = (float*)alloc((size_t)MP_ * HID_ * 4);
  bf16*  g1b   = (bf16*)alloc((size_t)MP_ * HID_ * 2);
  bf16*  h_bf  = (bf16*)alloc((size_t)MP_ * HID_ * 2);
  float* xw    = (float*)alloc((size_t)MP_ * HID_ * 4);
  float* agg   = (float*)alloc((size_t)MP_ * HID_ * 4);
  float* dinv  = (float*)alloc((size_t)MP_ * 4);

  auto cdiv = [](int a, int b) { return (a + b - 1) / b; };

  // precision prep (zero the A-pad rows once so all downstream values are defined)
  eg_f2bf<<<cdiv(NN_ * ND_, 256), 256, 0, stream>>>(nf, nf_bf, NN_ * ND_);
  eg_zero_bf<<<cdiv((MP_ - NN_) * ND_, 256), 256, 0, stream>>>(nf_bf + (size_t)NN_ * ND_,
                                                               (MP_ - NN_) * ND_);
  eg_transpose_bf<<<cdiv(ND_ * HID_, 256), 256, 0, stream>>>(enc_w1, w1T, ND_, HID_);
  eg_transpose_bf<<<cdiv(HID_ * HID_, 256), 256, 0, stream>>>(enc_w2, w2T, HID_, HID_);
  for (int l = 0; l < NL_; ++l)
    eg_transpose_bf<<<cdiv(HID_ * HID_, 256), 256, 0, stream>>>(
        gnn_w + (size_t)l * HID_ * HID_, gT + (size_t)l * HID_ * HID_, HID_, HID_);
  eg_transpose_bf<<<cdiv(2 * HID_ * HID_, 256), 256, 0, stream>>>(edge_w1, e1T, 2 * HID_, HID_);

  dim3 gemm_grid(HID_ / 64, MP_ / 64);
  // node encoder
  eg_wmma_gemm<ND_,  true, true,  false><<<gemm_grid, 128, 0, stream>>>(nf_bf, w1T, enc_b1, t1, nullptr);
  eg_ln_silu<<<MP_ / 8, 256, 0, stream>>>(t1, enc_lng, enc_lnb, g1b, MP_);
  eg_wmma_gemm<HID_, true, false, true ><<<gemm_grid, 128, 0, stream>>>(g1b, w2T, enc_b2, nullptr, h_bf);

  // GCN degree norm (self-loops included; pad nodes get deg=1 -> dinv=1)
  eg_fill1<<<cdiv(MP_, 256), 256, 0, stream>>>(dinv, MP_);
  eg_deg<<<cdiv(NE_, 256), 256, 0, stream>>>(dst, dinv, NE_);
  eg_rsqrt<<<cdiv(MP_, 256), 256, 0, stream>>>(dinv, MP_);

  // GCN layers
  for (int l = 0; l < NL_; ++l) {
    eg_wmma_gemm<HID_, false, true, false><<<gemm_grid, 128, 0, stream>>>(
        h_bf, gT + (size_t)l * HID_ * HID_, nullptr, xw, nullptr);
    eg_agg_init<<<cdiv(MP_ * HID_, 256), 256, 0, stream>>>(xw, dinv, agg, MP_ * HID_);
    eg_agg_edges<<<cdiv(NE_ * 32, 256), 256, 0, stream>>>(xw, src, dst, dinv, agg, NE_);
    eg_finalize<<<cdiv(MP_ * HID_, 256), 256, 0, stream>>>(agg, gnn_b + (size_t)l * HID_, h_bf, MP_ * HID_);
  }

  // fused edge predictor
  eg_edge_head<<<NE_ / 16, 256, 0, stream>>>(h_bf, src, dst, e1T, edge_b1,
                                             e_lng, e_lnb, edge_w2, edge_b2, out);
}
